// Attention_47605417509340
// MI455X (gfx1250) — matrix operand order repaired and checked by reading, not compiled, not run on hardware
//
#include <hip/hip_runtime.h>
#include <hip/hip_fp16.h>

typedef _Float16 half_t;
typedef __attribute__((ext_vector_type(16))) _Float16 v16h;
typedef __attribute__((ext_vector_type(8)))  _Float16 v8h;
typedef __attribute__((ext_vector_type(8)))  float    v8f;
typedef __attribute__((ext_vector_type(4)))  unsigned int v4u;
typedef __attribute__((ext_vector_type(8)))  int      v8i;
typedef __attribute__((ext_vector_type(4)))  int      v4i;

#define HIDDEN 4096
#define SEQ    2048
#define BATCH  2
#define NH     32
#define NKV    8
#define HD     128
#define TOKENS (BATCH * SEQ)

#if defined(__has_builtin)
#if __has_builtin(__builtin_amdgcn_tensor_load_to_lds) && __has_builtin(__builtin_amdgcn_s_wait_tensorcnt)
#define USE_TDM 1
#endif
#endif

// ---------------------------------------------------------------------------
// fp32 -> fp16 conversion
// ---------------------------------------------------------------------------
__global__ __launch_bounds__(256) void f32_to_f16_kernel(const float* __restrict__ in,
                                                         half_t* __restrict__ out, int n) {
  int i = blockIdx.x * 256 + threadIdx.x;
  if (i < n) out[i] = (half_t)in[i];
}

// ---------------------------------------------------------------------------
// RoPE, in place on fp16 Q or K:   X layout [TOKENS][n_heads*HD]
// ---------------------------------------------------------------------------
__global__ __launch_bounds__(256) void rope_kernel(half_t* __restrict__ X,
                                                   const int* __restrict__ pos,
                                                   int n_heads) {
  long long idx   = (long long)blockIdx.x * 256 + threadIdx.x;
  long long total = (long long)TOKENS * n_heads * (HD / 2);
  if (idx >= total) return;
  int d       = (int)(idx % (HD / 2));
  long long t = idx / (HD / 2);
  int h       = (int)(t % n_heads);
  long long tok = t / n_heads;

  float p    = (float)pos[tok % SEQ];
  float freq = __powf(10000.0f, -(float)(2 * d) / (float)HD);
  float s, c;
  __sincosf(p * freq, &s, &c);

  half_t* ptr = X + tok * (long long)(n_heads * HD) + h * HD + d;
  float v0 = (float)ptr[0];
  float v1 = (float)ptr[HD / 2];
  ptr[0]      = (half_t)(v0 * c - v1 * s);
  ptr[HD / 2] = (half_t)(v1 * c + v0 * s);
}

// ---------------------------------------------------------------------------
// WMMA GEMM:  C[M,N] = A[M,K] * B[N,K]^T   (A,B fp16 row-major, C fp16/fp32)
// 256 threads = 8 waves (2x4); wave tile 64x64 (4x4 wmma), block tile 128x256.
// 16 WMMAs per 8 fragment loads per K-step of 32.
// ---------------------------------------------------------------------------
template <typename OutT>
__global__ __launch_bounds__(256) void gemm_wmma_kernel(const half_t* __restrict__ A,
                                                        const half_t* __restrict__ B,
                                                        OutT* __restrict__ C,
                                                        int M, int N, int K) {
  const int lane = threadIdx.x & 31;
  const int wave = threadIdx.x >> 5;
  const int l16  = lane & 15;
  const int hf   = lane >> 4;
  const int wrow = wave >> 2;           // 0..1
  const int wcol = wave & 3;            // 0..3
  const int m0 = blockIdx.y * 128 + wrow * 64;
  const int n0 = blockIdx.x * 256 + wcol * 64;

  v8f acc[4][4] = {};

  const half_t* ar[4];
  const half_t* br[4];
#pragma unroll
  for (int i = 0; i < 4; ++i) {
    ar[i] = A + (size_t)(m0 + i * 16 + l16) * K;
    br[i] = B + (size_t)(n0 + i * 16 + l16) * K;
  }

  for (int k0 = 0; k0 < K; k0 += 32) {
    const int kc = k0 + hf * 16;
    v16h af[4], bf[4];
#pragma unroll
    for (int i = 0; i < 4; ++i) {
      af[i] = *(const v16h*)(ar[i] + kc);
      bf[i] = *(const v16h*)(br[i] + kc);
    }
    __builtin_prefetch((const void*)(ar[0] + kc + 32), 0, 1);
    __builtin_prefetch((const void*)(br[0] + kc + 32), 0, 1);
#pragma unroll
    for (int i = 0; i < 4; ++i)
#pragma unroll
      for (int j = 0; j < 4; ++j)
        acc[i][j] = __builtin_amdgcn_wmma_f32_16x16x32_f16(false, af[i], false, bf[j],
                                                           (short)0, acc[i][j], false, false);
  }

  // C/D layout: VGPR r, lane -> row = r + 8*(lane/16), col = lane%16
#pragma unroll
  for (int i = 0; i < 4; ++i)
#pragma unroll
    for (int j = 0; j < 4; ++j)
#pragma unroll
      for (int r = 0; r < 8; ++r) {
        int row = m0 + i * 16 + r + 8 * hf;
        int col = n0 + j * 16 + l16;
        C[(size_t)row * N + col] = (OutT)acc[i][j][r];
      }
}

// ---------------------------------------------------------------------------
// TDM: issue one tensor_load_to_lds moving a [32 x 128]-half tile (row stride
// NKV*HD halfs) from global into LDS.  Descriptor per CDNA5 ISA D# groups.
// ---------------------------------------------------------------------------
#ifdef USE_TDM
__device__ inline void tdm_load_tile_2d(unsigned lds_addr, const half_t* gptr) {
  unsigned long long ga = (unsigned long long)gptr;
  v4u g0;
  g0[0] = 1u;                                            // count=1, user mode
  g0[1] = lds_addr;                                      // lds_addr[63:32]
  g0[2] = (unsigned)(ga & 0xFFFFFFFFu);                  // global_addr[95:64]
  g0[3] = (unsigned)((ga >> 32) & 0x1FFFFFFu)            // global_addr[120:96]
        | (2u << 30);                                    // type=2 ("image")
  v8i g1;
  g1[0] = 0x10000;                                       // data_size=1 (2 bytes)
  g1[1] = (int)((unsigned)HD << 16);                     // tensor_dim0 = 128
  g1[2] = (int)((unsigned)(SEQ & 0xFFFF) << 16);         // tensor_dim1 = 2048
  g1[3] = (int)((unsigned)HD << 16);                     // tile_dim0 = 128
  g1[4] = 32;                                            // tile_dim1 = 32
  g1[5] = NKV * HD;                                      // tensor_dim0_stride = 1024
  g1[6] = 0;
  g1[7] = 0;
  v4i z4 = {0, 0, 0, 0};
  v8i z8 = {0, 0, 0, 0, 0, 0, 0, 0};
  // 6-arg form (clang-23 / therock headers): (g0, g1, g2, g3, g4, cpol)
  __builtin_amdgcn_tensor_load_to_lds(g0, g1, z4, z4, z8, 0);
}
#endif

// ---------------------------------------------------------------------------
// Flash attention (GQA 4:1), fp16 in/out, fp32 online softmax.
// grid = (SEQ/128, BATCH*NH); block = 256 (8 waves, 16 q-rows per wave).
// K/V tiles (32 keys x 128 dims) staged into LDS by the Tensor Data Mover
// (fallback: per-lane global_load_async_to_lds_b128); K B-fragments read
// back with plain ds_load_b128 (one key row per lane, no transpose).
// ---------------------------------------------------------------------------
__global__ __launch_bounds__(256) void flash_attn_kernel(const half_t* __restrict__ Q,
                                                         const half_t* __restrict__ K,
                                                         const half_t* __restrict__ V,
                                                         half_t* __restrict__ O) {
  __shared__ half_t Kn[32][HD];        // [key][dim]   8 KB
  __shared__ half_t Vt[32][HD];        // [key][dim]   8 KB
  __shared__ half_t Pst[8][16][32];    // per-wave P staging  8 KB

  const int lane = threadIdx.x & 31;
  const int wave = threadIdx.x >> 5;
  const int l16  = lane & 15;
  const int hf   = lane >> 4;

  const int bh  = blockIdx.y;
  const int b   = bh >> 5;
  const int h   = bh & 31;
  const int kvh = h >> 2;

  // Load this wave's 16x128 Q tile as 4 A-fragments (K-dim steps of 32)
  const int qrow = blockIdx.x * 128 + wave * 16 + l16;
  const size_t qoff = ((size_t)(b * SEQ + qrow)) * (NH * HD) + h * HD;
  v16h qf[4];
#pragma unroll
  for (int kd = 0; kd < 4; ++kd)
    qf[kd] = *(const v16h*)(Q + qoff + kd * 32 + hf * 16);

  v8f o[8] = {};
  float mr[8], lr[8];
#pragma unroll
  for (int r = 0; r < 8; ++r) { mr[r] = -3.0e38f; lr[r] = 0.0f; }

  const half_t* Kb = K + (size_t)b * SEQ * (NKV * HD) + kvh * HD;
  const half_t* Vb = V + (size_t)b * SEQ * (NKV * HD) + kvh * HD;
  const float scale = 0.08838834764831845f;   // 1/sqrt(128)

  for (int ks = 0; ks < SEQ; ks += 32) {
#ifdef USE_TDM
    // ---- TDM-stage K and V tiles: 2 descriptor-driven DMAs from wave 0 ----
    if (wave == 0) {
      unsigned kl = (unsigned)(unsigned long long)&Kn[0][0];
      unsigned vl = (unsigned)(unsigned long long)&Vt[0][0];
      tdm_load_tile_2d(kl, Kb + (size_t)ks * (NKV * HD));
      tdm_load_tile_2d(vl, Vb + (size_t)ks * (NKV * HD));
      __builtin_amdgcn_s_wait_tensorcnt(0);
    }
#else
    // ---- async-stage K and V tiles: 2x (2 chunks of 16B) per thread ----
    {
#pragma unroll
      for (int rep = 0; rep < 2; ++rep) {
        int chunk = threadIdx.x + rep * 256;     // 0..511
        int key   = chunk >> 4;                  // 32 keys, 16 chunks per key
        int d0    = (chunk & 15) * 8;            // halfs
        unsigned koff = (unsigned)(unsigned long long)&Kn[key][d0];
        unsigned voff = (unsigned)(unsigned long long)&Vt[key][d0];
        unsigned long long kga = (unsigned long long)(Kb + (size_t)(ks + key) * (NKV * HD) + d0);
        unsigned long long vga = (unsigned long long)(Vb + (size_t)(ks + key) * (NKV * HD) + d0);
        asm volatile("global_load_async_to_lds_b128 %0, %1, off"
                     :: "v"(koff), "v"(kga) : "memory");
        asm volatile("global_load_async_to_lds_b128 %0, %1, off"
                     :: "v"(voff), "v"(vga) : "memory");
      }
      asm volatile("s_wait_asynccnt 0x0" ::: "memory");
    }
#endif
    __syncthreads();

    // ---- scores: S(16x32) = Q(16x128) * K^T, 2 col-tiles x 4 K-steps ----
    v8f sc[2];
#pragma unroll
    for (int c = 0; c < 2; ++c) {
      v8f a = {};
#pragma unroll
      for (int kd = 0; kd < 4; ++kd) {
        // read a 16x32 slab of K out of LDS as a B fragment (key row per lane, dims 16*hf..+15)
        unsigned kaddr = (unsigned)(unsigned long long)&Kn[c * 16 + l16][kd * 32 + hf * 16];
        v8h t0, t1;
        asm volatile("ds_load_b128 %0, %1"           : "=v"(t0) : "v"(kaddr) : "memory");
        asm volatile("ds_load_b128 %0, %1 offset:16" : "=v"(t1) : "v"(kaddr) : "memory");
        v16h bf;
#pragma unroll
        for (int j = 0; j < 8; ++j) { bf[j] = t0[j]; bf[j + 8] = t1[j]; }
        a = __builtin_amdgcn_wmma_f32_16x16x32_f16(false, qf[kd], false, bf, (short)0, a, false, false);
      }
#pragma unroll
      for (int r = 0; r < 8; ++r) a[r] *= scale;
      sc[c] = a;
    }

    // ---- online softmax (rows live in 16-lane halves of the wave) ----
#pragma unroll
    for (int r = 0; r < 8; ++r) {
      float v = fmaxf(sc[0][r], sc[1][r]);
      v = fmaxf(v, __shfl_xor(v, 1));
      v = fmaxf(v, __shfl_xor(v, 2));
      v = fmaxf(v, __shfl_xor(v, 4));
      v = fmaxf(v, __shfl_xor(v, 8));
      float mn    = fmaxf(mr[r], v);
      float alpha = __expf(mr[r] - mn);
      mr[r] = mn;
      lr[r] *= alpha;
#pragma unroll
      for (int j = 0; j < 8; ++j) o[j][r] *= alpha;

      float ps = 0.0f;
#pragma unroll
      for (int c = 0; c < 2; ++c) {
        float p = __expf(sc[c][r] - mn);
        ps += p;
        Pst[wave][r + 8 * hf][c * 16 + l16] = (half_t)p;
      }
      ps += __shfl_xor(ps, 1);
      ps += __shfl_xor(ps, 2);
      ps += __shfl_xor(ps, 4);
      ps += __shfl_xor(ps, 8);
      lr[r] += ps;
    }

    // ---- P*V: reload P as A-fragment, 8 wmma over the 128-wide head dim ----
    v16h pf = *(const v16h*)&Pst[wave][l16][hf * 16];
#pragma unroll
    for (int j = 0; j < 8; ++j) {
      v16h vf;
#pragma unroll
      for (int s = 0; s < 16; ++s) vf[s] = Vt[hf * 16 + s][j * 16 + l16];
      o[j] = __builtin_amdgcn_wmma_f32_16x16x32_f16(false, pf, false, vf, (short)0, o[j], false, false);
    }
    __syncthreads();
  }

  // ---- epilogue: normalize and store fp16 ----
#pragma unroll
  for (int r = 0; r < 8; ++r) {
    float inv = 1.0f / lr[r];
    int row   = blockIdx.x * 128 + wave * 16 + r + 8 * hf;
    size_t base = ((size_t)(b * SEQ + row)) * (NH * HD) + h * HD;
#pragma unroll
    for (int j = 0; j < 8; ++j)
      O[base + j * 16 + l16] = (half_t)(o[j][r] * inv);
  }
}

// ---------------------------------------------------------------------------
// Host orchestration
// ---------------------------------------------------------------------------
extern "C" void kernel_launch(void* const* d_in, const int* in_sizes, int n_in,
                              void* d_out, int out_size, void* d_ws, size_t ws_size,
                              hipStream_t stream) {
  const float* X   = (const float*)d_in[0];
  const int*   pos = (const int*)  d_in[1];
  const float* Wq  = (const float*)d_in[2];
  const float* Wk  = (const float*)d_in[3];
  const float* Wv  = (const float*)d_in[4];
  const float* Wo  = (const float*)d_in[5];
  float* out = (float*)d_out;

  // workspace layout (fp16 elements); X buffer is reused for attention output
  half_t* Xh  = (half_t*)d_ws;
  half_t* Wqh = Xh  + (size_t)TOKENS * HIDDEN;
  half_t* Wkh = Wqh + (size_t)HIDDEN * HIDDEN;
  half_t* Wvh = Wkh + (size_t)(NKV * HD) * HIDDEN;
  half_t* Woh = Wvh + (size_t)(NKV * HD) * HIDDEN;
  half_t* Qh  = Woh + (size_t)HIDDEN * HIDDEN;
  half_t* Kh  = Qh  + (size_t)TOKENS * HIDDEN;
  half_t* Vh  = Kh  + (size_t)TOKENS * (NKV * HD);
  half_t* Ah  = Xh;   // alias: X is dead after the V projection

  // 1) convert activations + weights to fp16
  {
    int n;
    n = TOKENS * HIDDEN;   f32_to_f16_kernel<<<(n + 255) / 256, 256, 0, stream>>>(X,  Xh,  n);
    n = HIDDEN * HIDDEN;   f32_to_f16_kernel<<<(n + 255) / 256, 256, 0, stream>>>(Wq, Wqh, n);
    n = NKV * HD * HIDDEN; f32_to_f16_kernel<<<(n + 255) / 256, 256, 0, stream>>>(Wk, Wkh, n);
    n = NKV * HD * HIDDEN; f32_to_f16_kernel<<<(n + 255) / 256, 256, 0, stream>>>(Wv, Wvh, n);
    n = HIDDEN * HIDDEN;   f32_to_f16_kernel<<<(n + 255) / 256, 256, 0, stream>>>(Wo, Woh, n);
  }

  // 2) QKV projections (WMMA GEMM, C = X * W^T); block tile 128x256
  gemm_wmma_kernel<half_t><<<dim3(HIDDEN / 256,     TOKENS / 128), 256, 0, stream>>>(Xh, Wqh, Qh, TOKENS, HIDDEN,   HIDDEN);
  gemm_wmma_kernel<half_t><<<dim3((NKV * HD) / 256, TOKENS / 128), 256, 0, stream>>>(Xh, Wkh, Kh, TOKENS, NKV * HD, HIDDEN);
  gemm_wmma_kernel<half_t><<<dim3((NKV * HD) / 256, TOKENS / 128), 256, 0, stream>>>(Xh, Wvh, Vh, TOKENS, NKV * HD, HIDDEN);

  // 3) RoPE on Q and K
  {
    long long nq = (long long)TOKENS * NH  * (HD / 2);
    long long nk = (long long)TOKENS * NKV * (HD / 2);
    rope_kernel<<<(unsigned)((nq + 255) / 256), 256, 0, stream>>>(Qh, pos, NH);
    rope_kernel<<<(unsigned)((nk + 255) / 256), 256, 0, stream>>>(Kh, pos, NKV);
  }

  // 4) flash attention (GQA)
  flash_attn_kernel<<<dim3(SEQ / 128, BATCH * NH), 256, 0, stream>>>(Qh, Kh, Vh, Ah);

  // 5) output projection, fp32 result straight to d_out
  gemm_wmma_kernel<float><<<dim3(HIDDEN / 256, TOKENS / 128), 256, 0, stream>>>(Ah, Woh, out, TOKENS, HIDDEN, HIDDEN);

  (void)in_sizes; (void)n_in; (void)out_size; (void)ws_size;
}
